// IDMPNN_Global_parallel_2559800509140
// MI455X (gfx1250) — matrix operand order repaired and checked by reading, not compiled, not run on hardware
//
#include <hip/hip_runtime.h>
#include <hip/hip_bf16.h>

// MI455X / gfx1250, wave32. All GEMMs via v_wmma_f32_16x16x32_bf16.
typedef __bf16 bf16_t;
typedef __attribute__((ext_vector_type(16))) __bf16 v16bf;
typedef __attribute__((ext_vector_type(8)))  __bf16 v8bf;
typedef __attribute__((ext_vector_type(8)))  float  v8f;

#define HD 128   // hidden dim, also K-dim of every GEMM

// ---------------------------------------------------------------------------
// f32 -> bf16 conversion (adjacency)
// ---------------------------------------------------------------------------
__global__ void cvt_f2bf_kernel(const float* __restrict__ in, bf16_t* __restrict__ out, long n) {
  long i = (long)blockIdx.x * 256 + threadIdx.x;
  if (i < n) out[i] = (bf16_t)in[i];
}

// f32 (mats x 128 x 128, row-major KxN) -> bf16 transposed (mats x N x K)
__global__ void cvtT_f2bf_kernel(const float* __restrict__ in, bf16_t* __restrict__ out, long n) {
  long idx = (long)blockIdx.x * 256 + threadIdx.x;
  if (idx >= n) return;
  const int d   = (int)(idx & 16383);
  const long mat = idx >> 14;
  const int nn = d >> 7, kk = d & 127;
  out[(mat << 14) + (nn << 7) + kk] = (bf16_t)in[(mat << 14) + (kk << 7) + nn];
}

// ---------------------------------------------------------------------------
// WMMA GEMM: C(MxN, bf16) = [relu]( A(MxK=128, bf16) @ B(128xN, bf16) + bias )
// BT=true : B is pre-transposed (NxK row-major, weights) -> pure global b128
//           loads, no LDS, no barriers.
// BT=false: B row-major KxN (activations); 32x16 tile staged transposed in
//           LDS so fragment reads are two 16-byte ds loads.
// Batched: A batch index = blockIdx.z / aDiv (lets sadj[s] = adj[s/16]).
// Block = 128 threads = 4 waves; tile = 64(M) x 16(N); wave owns 16x16 (v8f).
// ---------------------------------------------------------------------------
template<bool RELU, bool BT>
__global__ __launch_bounds__(128) void gemm_bf16_kernel(
    const bf16_t* __restrict__ A, long aStride, int aDiv, int lda,
    const bf16_t* __restrict__ Bw, long bStride, int ldb,
    const float* __restrict__ bias,
    bf16_t* __restrict__ C, long cStride, int ldc,
    int M, int N)
{
  const int bz = blockIdx.z;
  A  += (long)(bz / aDiv) * aStride;
  Bw += (long)bz * bStride;
  C  += (long)bz * cStride;

  const int wave = threadIdx.x >> 5;
  const int lane = threadIdx.x & 31;
  const int m0   = blockIdx.x * 64 + wave * 16;
  const int n0   = blockIdx.y * 16;
  const bool mv  = (m0 < M);           // M is always a multiple of 16
  const int hi   = lane >> 4;          // lane group (K-offset select)
  const int l15  = lane & 15;

  if (BT && !mv) return;               // barrier-free path can exit early

  v8f acc = {0.f,0.f,0.f,0.f,0.f,0.f,0.f,0.f};
  const bf16_t* arow = A + (long)(m0 + l15) * lda + hi * 8;

  if constexpr (BT) {
    // B^T: lane n = n0+l15 needs K = kk + hi*8 + [0..7] and kk + 16 + hi*8 + [0..7].
    const bf16_t* brow = Bw + (long)(n0 + l15) * HD + hi * 8;
    #pragma unroll
    for (int kk = 0; kk < HD; kk += 32) {
      v8bf a0 = *(const v8bf*)(arow + kk);
      v8bf a1 = *(const v8bf*)(arow + kk + 16);
      v8bf b0 = *(const v8bf*)(brow + kk);
      v8bf b1 = *(const v8bf*)(brow + kk + 16);
      v16bf fa, fb;
      #pragma unroll
      for (int e = 0; e < 8; ++e) {
        fa[e] = a0[e]; fa[e + 8] = a1[e];
        fb[e] = b0[e]; fb[e + 8] = b1[e];
      }
      acc = __builtin_amdgcn_wmma_f32_16x16x32_bf16(
          false, fa, false, fb, (short)0, acc, false, false);
    }
  } else {
    __shared__ __align__(16) bf16_t Bs[16][40];   // [n][k], padded rows
    for (int kk = 0; kk < HD; kk += 32) {
      // cooperative transpose-stage: thread t loads 4 contiguous n for one k
      {
        const int t4  = threadIdx.x * 4;
        const int row = t4 >> 4;        // k within tile (0..31)
        const int col = t4 & 15;        // n within tile (0,4,8,12)
        const bf16_t* src = Bw + (long)(kk + row) * ldb + n0 + col;
        Bs[col + 0][row] = src[0];
        Bs[col + 1][row] = src[1];
        Bs[col + 2][row] = src[2];
        Bs[col + 3][row] = src[3];
      }
      __syncthreads();
      if (mv) {
        v8bf a0 = *(const v8bf*)(arow + kk);
        v8bf a1 = *(const v8bf*)(arow + kk + 16);
        v8bf b0 = *(const v8bf*)&Bs[l15][hi * 8];
        v8bf b1 = *(const v8bf*)&Bs[l15][hi * 8 + 16];
        v16bf fa, fb;
        #pragma unroll
        for (int e = 0; e < 8; ++e) {
          fa[e] = a0[e]; fa[e + 8] = a1[e];
          fb[e] = b0[e]; fb[e + 8] = b1[e];
        }
        acc = __builtin_amdgcn_wmma_f32_16x16x32_bf16(
            false, fa, false, fb, (short)0, acc, false, false);
      }
      __syncthreads();
    }
  }

  if (mv) {
    const int n = n0 + l15;
    const float bv = bias ? bias[n] : 0.f;
    #pragma unroll
    for (int r = 0; r < 8; ++r) {
      const int m = m0 + r + hi * 8;           // C layout: VGPR r -> M=r / 8+r
      float v = acc[r] + bv;
      if (RELU) v = fmaxf(v, 0.f);
      C[(long)m * ldc + n] = (bf16_t)v;
    }
  }
}

// ---------------------------------------------------------------------------
// out = [res +] relu( LN(t) * g + b )    rows of length 128, 1 wave / row
// ---------------------------------------------------------------------------
template<bool RES_BF16, bool OUT_BF16>
__global__ __launch_bounds__(256) void ln_relu_res_kernel(
    const bf16_t* __restrict__ T, const void* __restrict__ Res, void* __restrict__ Out,
    const float* __restrict__ gam, const float* __restrict__ bet, long rows)
{
  const long row = (long)blockIdx.x * 8 + (threadIdx.x >> 5);
  if (row >= rows) return;
  const int lane = threadIdx.x & 31;
  const bf16_t* t = T + (row << 7);
  float v0 = (float)t[lane], v1 = (float)t[lane + 32],
        v2 = (float)t[lane + 64], v3 = (float)t[lane + 96];
  float s = v0 + v1 + v2 + v3;
  #pragma unroll
  for (int o = 16; o > 0; o >>= 1) s += __shfl_xor(s, o, 32);
  const float mean = s * 0.0078125f;
  const float d0 = v0 - mean, d1 = v1 - mean, d2 = v2 - mean, d3 = v3 - mean;
  float q = d0*d0 + d1*d1 + d2*d2 + d3*d3;
  #pragma unroll
  for (int o = 16; o > 0; o >>= 1) q += __shfl_xor(q, o, 32);
  const float inv = rsqrtf(q * 0.0078125f + 1e-5f);
  const float dv[4] = {d0, d1, d2, d3};
  #pragma unroll
  for (int i = 0; i < 4; ++i) {
    const int c = lane + 32 * i;
    float y = fmaxf(dv[i] * inv * gam[c] + bet[c], 0.f);
    float r = 0.f;
    if (Res) r = RES_BF16 ? (float)((const bf16_t*)Res)[(row << 7) + c]
                          : ((const float*)Res)[(row << 7) + c];
    const float ov = r + y;
    if (OUT_BF16) ((bf16_t*)Out)[(row << 7) + c] = (bf16_t)ov;
    else          ((float*) Out)[(row << 7) + c] = ov;
  }
}

// ---------------------------------------------------------------------------
// h1 = relu(x * in_W1 + in_b1)   (rank-1 input embedding), (4096,128) bf16
// ---------------------------------------------------------------------------
__global__ void in_embed_kernel(const float* __restrict__ x, const float* __restrict__ W1,
                                const float* __restrict__ b1, bf16_t* __restrict__ h1) {
  const int idx = blockIdx.x * 256 + threadIdx.x;     // 524288
  const int c = idx & 127, r = idx >> 7;
  h1[idx] = (bf16_t)fmaxf(x[r] * W1[c] + b1[c], 0.f);
}

// ---------------------------------------------------------------------------
// z(Ssub=512, N=128, P=6, H=128) init: ones, labeled nodes take idemb[allperm]
// ---------------------------------------------------------------------------
__global__ void z_init_kernel(bf16_t* __restrict__ z, const int* __restrict__ subgs,
                              const int* __restrict__ allperm, const float* __restrict__ idemb) {
  const long idx = (long)blockIdx.x * 256 + threadIdx.x;   // 50,331,648
  const int d = (int)(idx & 127);
  const long r = idx >> 7;               // (s,n,p), p fastest
  const int p = (int)(r % 6);
  const long sn = r / 6;
  const int n = (int)(sn & 127);
  const int s = (int)(sn >> 7);
  float val = 1.f;
  const int* sg = subgs + s * 4;
  #pragma unroll
  for (int j = 0; j < 4; ++j)
    if (sg[j] == n) val = idemb[allperm[j * 6 + p] * 128 + d];
  z[idx] = (bf16_t)val;
}

// zm[s,n,d] = mask(n) * mean_p z[s,n,p,d]
__global__ void z_mean_mask_kernel(const bf16_t* __restrict__ z, const int* __restrict__ num_node,
                                   bf16_t* __restrict__ zm) {
  const long idx = (long)blockIdx.x * 256 + threadIdx.x;   // 8,388,608
  const int d = (int)(idx & 127);
  const int n = (int)((idx >> 7) & 127);
  const int s = (int)(idx >> 14);
  const bf16_t* base = z + ((((long)s * 128 + n) * 6) << 7) + d;
  float acc = 0.f;
  #pragma unroll
  for (int p = 0; p < 6; ++p) acc += (float)base[p << 7];
  acc = (n >= num_node[s >> 4]) ? 0.f : acc * (1.f / 6.f);
  zm[idx] = (bf16_t)acc;
}

// hs[s,n,:] = h[s/16,n,:] + t1[s,n,:]
__global__ void hs_gather_add_kernel(const bf16_t* __restrict__ h, const bf16_t* __restrict__ t1,
                                     bf16_t* __restrict__ hs) {
  const long idx = (long)blockIdx.x * 256 + threadIdx.x;   // 8,388,608
  const int d = (int)(idx & 127);
  const int n = (int)((idx >> 7) & 127);
  const int g = (int)(idx >> 14) >> 4;
  hs[idx] = (bf16_t)((float)h[(((long)g * 128 + n) << 7) + d] + (float)t1[idx]);
}

// q[s,d] = sum_n hs[s,n,d]
__global__ void colsum_kernel(const bf16_t* __restrict__ hs, bf16_t* __restrict__ q) {
  const int idx = blockIdx.x * 256 + threadIdx.x;          // 65536
  const int s = idx >> 7, d = idx & 127;
  float acc = 0.f;
  for (int n = 0; n < 128; ++n) acc += (float)hs[(((long)s * 128 + n) << 7) + d];
  q[idx] = (bf16_t)acc;
}

// pooled[g,d] = max over 16 subgraphs of hs2[g*16+s, d]
__global__ void segmax_kernel(const bf16_t* __restrict__ hs2,
                              float* __restrict__ pf, bf16_t* __restrict__ pb) {
  const int idx = blockIdx.x * 256 + threadIdx.x;          // 4096
  if (idx >= 32 * 128) return;
  const int g = idx >> 7, d = idx & 127;
  float m = -3.4e38f;
  for (int s = 0; s < 16; ++s)
    m = fmaxf(m, (float)hs2[((g * 16 + s) << 7) + d]);
  pf[idx] = m;
  pb[idx] = (bf16_t)m;
}

// out[g] = pooled2[g,:] . out_W + out_b      (one wave per graph)
__global__ void out_dot_kernel(const float* __restrict__ P, const float* __restrict__ W,
                               const float* __restrict__ b, float* __restrict__ out) {
  const int g = blockIdx.x, lane = threadIdx.x;
  float s = 0.f;
  #pragma unroll
  for (int i = 0; i < 4; ++i) s += P[g * 128 + lane + 32 * i] * W[lane + 32 * i];
  #pragma unroll
  for (int o = 16; o > 0; o >>= 1) s += __shfl_xor(s, o, 32);
  if (lane == 0) out[g] = s + b[0];
}

// ---------------------------------------------------------------------------
extern "C" void kernel_launch(void* const* d_in, const int* in_sizes, int n_in,
                              void* d_out, int out_size, void* d_ws, size_t ws_size,
                              hipStream_t stream) {
  (void)in_sizes; (void)n_in; (void)out_size; (void)ws_size;
  const float* x     = (const float*)d_in[0];
  const float* adj   = (const float*)d_in[1];
  const float* idemb = (const float*)d_in[2];
  const float* in_W1 = (const float*)d_in[3];
  const float* in_b1 = (const float*)d_in[4];
  const float* in_W2 = (const float*)d_in[5];
  const float* in_b2 = (const float*)d_in[6];
  const float* in_g  = (const float*)d_in[7];
  const float* in_bn = (const float*)d_in[8];
  const float* gW1 = (const float*)d_in[9];
  const float* gb1 = (const float*)d_in[10];
  const float* gW2 = (const float*)d_in[11];
  const float* gb2 = (const float*)d_in[12];
  const float* gg  = (const float*)d_in[13];
  const float* gbn = (const float*)d_in[14];
  const float* iW1 = (const float*)d_in[15];
  const float* ib1 = (const float*)d_in[16];
  const float* iW2 = (const float*)d_in[17];
  const float* ib2 = (const float*)d_in[18];
  const float* ig  = (const float*)d_in[19];
  const float* ibn = (const float*)d_in[20];
  const float* glW1 = (const float*)d_in[21];
  const float* glb1 = (const float*)d_in[22];
  const float* glW2 = (const float*)d_in[23];
  const float* glb2 = (const float*)d_in[24];
  const float* glg  = (const float*)d_in[25];
  const float* glbn = (const float*)d_in[26];
  const float* s1_W = (const float*)d_in[27];
  const float* s1_b = (const float*)d_in[28];
  const float* s2_W = (const float*)d_in[29];
  const float* s2_b = (const float*)d_in[30];
  const float* rW1 = (const float*)d_in[31];
  const float* rb1 = (const float*)d_in[32];
  const float* rW2 = (const float*)d_in[33];
  const float* rb2 = (const float*)d_in[34];
  const float* rg  = (const float*)d_in[35];
  const float* rbn = (const float*)d_in[36];
  const float* out_W = (const float*)d_in[37];
  const float* out_b = (const float*)d_in[38];
  const int* subgs    = (const int*)d_in[39];
  const int* num_node = (const int*)d_in[41];
  const int* allperm  = (const int*)d_in[42];
  float* outp = (float*)d_out;

  // -------- workspace carve-out (deterministic) --------
  char* ws = (char*)d_ws;
  size_t off = 0;
  auto alloc = [&](size_t bytes) -> char* {
    char* p = ws + off;
    off += (bytes + 255) & ~((size_t)255);
    return p;
  };
  const long ZN = (long)512 * 128 * 6 * 128;             // 50,331,648
  bf16_t* adjbf = (bf16_t*)alloc((size_t)32 * 128 * 128 * 2);
  bf16_t* h     = (bf16_t*)alloc((size_t)4096 * 128 * 2);
  bf16_t* z     = (bf16_t*)alloc((size_t)ZN * 2);
  bf16_t* bufA  = (bf16_t*)alloc((size_t)ZN * 2);
  bf16_t* bufB  = (bf16_t*)alloc((size_t)ZN * 2);
  bf16_t* hs    = (bf16_t*)alloc((size_t)512 * 128 * 128 * 2);
  bf16_t* q     = (bf16_t*)alloc((size_t)512 * 128 * 2);
  bf16_t* hs2   = (bf16_t*)alloc((size_t)512 * 128 * 2);
  float*  pooledf  = (float*) alloc((size_t)32 * 128 * 4);
  bf16_t* pooledbf = (bf16_t*)alloc((size_t)32 * 128 * 2);
  float*  pooled2f = (float*) alloc((size_t)32 * 128 * 4);

  // weights: convert + transpose (stored N x K for barrier-free B fragments)
  auto cvtwT = [&](const float* src, long n) -> bf16_t* {
    bf16_t* dst = (bf16_t*)alloc((size_t)n * 2);
    cvtT_f2bf_kernel<<<dim3((unsigned)((n + 255) / 256)), dim3(256), 0, stream>>>(src, dst, n);
    return dst;
  };
  cvt_f2bf_kernel<<<dim3(2048), dim3(256), 0, stream>>>(adj, adjbf, (long)32 * 128 * 128);
  bf16_t* inW2bf = cvtwT(in_W2, 16384);
  bf16_t* gW1bf  = cvtwT(gW1, 3 * 16384);
  bf16_t* gW2bf  = cvtwT(gW2, 3 * 16384);
  bf16_t* iW1bf  = cvtwT(iW1, 2 * 16384);
  bf16_t* iW2bf  = cvtwT(iW2, 2 * 16384);
  bf16_t* glW1bf = cvtwT(glW1, 2 * 16384);
  bf16_t* glW2bf = cvtwT(glW2, 2 * 16384);
  bf16_t* s1bf   = cvtwT(s1_W, 16384);
  bf16_t* s2bf   = cvtwT(s2_W, 16384);
  bf16_t* rW1bf  = cvtwT(rW1, 16384);
  bf16_t* rW2bf  = cvtwT(rW2, 16384);

  auto gemm = [&](const bf16_t* A, long aS, int aDiv, int lda,
                  const bf16_t* Bm, long bS, int ldb, const float* bias,
                  bf16_t* C, long cS, int ldc, int M, int N, int batches,
                  bool relu, bool bt) {
    dim3 g((unsigned)((M + 63) / 64), (unsigned)(N / 16), (unsigned)batches), blk(128);
    if (relu) {
      if (bt) gemm_bf16_kernel<true , true ><<<g, blk, 0, stream>>>(A, aS, aDiv, lda, Bm, bS, ldb, bias, C, cS, ldc, M, N);
      else    gemm_bf16_kernel<true , false><<<g, blk, 0, stream>>>(A, aS, aDiv, lda, Bm, bS, ldb, bias, C, cS, ldc, M, N);
    } else {
      if (bt) gemm_bf16_kernel<false, true ><<<g, blk, 0, stream>>>(A, aS, aDiv, lda, Bm, bS, ldb, bias, C, cS, ldc, M, N);
      else    gemm_bf16_kernel<false, false><<<g, blk, 0, stream>>>(A, aS, aDiv, lda, Bm, bS, ldb, bias, C, cS, ldc, M, N);
    }
  };
  auto ln_bf = [&](const bf16_t* T, const bf16_t* Res, bf16_t* Out,
                   const float* gam, const float* bet, long rows) {
    ln_relu_res_kernel<true, true><<<dim3((unsigned)((rows + 7) / 8)), dim3(256), 0, stream>>>(
        T, (const void*)Res, (void*)Out, gam, bet, rows);
  };

  // -------- input embedding: h = relu(LN(relu(x W1 + b1) W2 + b2)) --------
  in_embed_kernel<<<dim3(2048), dim3(256), 0, stream>>>(x, in_W1, in_b1, bufB);
  gemm(bufB, 0, 1, 128, inW2bf, 0, 128, in_b2, bufA, 0, 128, 4096, 128, 1, false, true);
  ln_bf(bufA, nullptr, h, in_g, in_bn, 4096);

  // -------- graph message passing (3 layers, batch=32) --------
  for (int l = 0; l < 3; ++l) {
    gemm(adjbf, 16384, 1, 128, h, 16384, 128, nullptr, bufA, 16384, 128, 128, 128, 32, false, false);
    gemm(bufA, 0, 1, 128, gW1bf + l * 16384, 0, 128, gb1 + l * 128, bufB, 0, 128, 4096, 128, 1, true, true);
    gemm(bufB, 0, 1, 128, gW2bf + l * 16384, 0, 128, gb2 + l * 128, bufA, 0, 128, 4096, 128, 1, false, true);
    ln_bf(bufA, h, h, gg + l * 128, gbn + l * 128, 4096);
  }

  // -------- identity labeling --------
  z_init_kernel<<<dim3(196608), dim3(256), 0, stream>>>(z, subgs, allperm, idemb);

  // -------- ID message passing (2 layers, batch=512, N=768) --------
  for (int l = 0; l < 2; ++l) {
    gemm(adjbf, 16384, 16, 128, z, 98304, 768, nullptr, bufA, 98304, 768, 128, 768, 512, false, false);
    gemm(bufA, 0, 1, 128, iW1bf + l * 16384, 0, 128, ib1 + l * 128, bufB, 0, 128, 393216, 128, 1, true, true);
    gemm(bufB, 0, 1, 128, iW2bf + l * 16384, 0, 128, ib2 + l * 128, bufA, 0, 128, 393216, 128, 1, false, true);
    ln_bf(bufA, z, z, ig + l * 128, ibn + l * 128, 393216);
  }

  // -------- setmlp1 + subgraph features --------
  z_mean_mask_kernel<<<dim3(32768), dim3(256), 0, stream>>>(z, num_node, bufB);
  gemm(bufB, 0, 1, 128, s1bf, 0, 128, s1_b, bufA, 0, 128, 65536, 128, 1, true, true);
  hs_gather_add_kernel<<<dim3(32768), dim3(256), 0, stream>>>(h, bufA, hs);

  // -------- global message passing (2 layers, batch=512) --------
  for (int l = 0; l < 2; ++l) {
    gemm(adjbf, 16384, 16, 128, hs, 16384, 128, nullptr, bufA, 16384, 128, 128, 128, 512, false, false);
    gemm(bufA, 0, 1, 128, glW1bf + l * 16384, 0, 128, glb1 + l * 128, bufB, 0, 128, 65536, 128, 1, true, true);
    gemm(bufB, 0, 1, 128, glW2bf + l * 16384, 0, 128, glb2 + l * 128, bufA, 0, 128, 65536, 128, 1, false, true);
    ln_bf(bufA, hs, hs, glg + l * 128, glbn + l * 128, 65536);
  }

  // -------- setmlp2 + segment max pool --------
  colsum_kernel<<<dim3(256), dim3(256), 0, stream>>>(hs, q);
  gemm(q, 0, 1, 128, s2bf, 0, 128, s2_b, hs2, 0, 128, 512, 128, 1, true, true);
  segmax_kernel<<<dim3(16), dim3(256), 0, stream>>>(hs2, pooledf, pooledbf);

  // -------- readout block + output head --------
  gemm(pooledbf, 0, 1, 128, rW1bf, 0, 128, rb1, bufB, 0, 128, 32, 128, 1, true, true);
  gemm(bufB, 0, 1, 128, rW2bf, 0, 128, rb2, bufA, 0, 128, 32, 128, 1, false, true);
  ln_relu_res_kernel<false, false><<<dim3(4), dim3(256), 0, stream>>>(
      bufA, (const void*)pooledf, (void*)pooled2f, rg, rbn, 32);
  out_dot_kernel<<<dim3(32), dim3(32), 0, stream>>>(pooled2f, out_W, out_b, outp);
}
